// GCNBottleneck_15659450761698
// MI455X (gfx1250) — matrix operand order repaired and checked by reading, not compiled, not run on hardware
//
#include <hip/hip_runtime.h>
#include <hip/hip_bf16.h>
#include <stdint.h>

#define BATCH 8
#define NPTS  4096
#define CDIM  256
#define HDIM  256
#define ODIM  128
#define KNN   8

typedef __attribute__((ext_vector_type(16))) __bf16 v16bf;
typedef __attribute__((ext_vector_type(8)))  float  v8f;
typedef __attribute__((ext_vector_type(4)))  unsigned int v4u;
typedef __attribute__((ext_vector_type(8)))  unsigned int v8u;

union Frag { v16bf v; unsigned int u[8]; };

__device__ __forceinline__ unsigned short f2bf(float f) {
  unsigned int u = __float_as_uint(f);
  u += 0x7FFFu + ((u >> 16) & 1u);          // round-to-nearest-even
  return (unsigned short)(u >> 16);
}
__device__ __forceinline__ float bf2f(unsigned short h) {
  return __uint_as_float(((unsigned int)h) << 16);
}

#define WMMA_BF16(A, B, C) \
  __builtin_amdgcn_wmma_f32_16x16x32_bf16(false, (A).v, false, (B).v, (short)0, (C), false, false)

// ---------------------------------------------------------------------------
// TDM: 1-D contiguous global->LDS DMA (bytes must be a multiple of 8).
// D# per CDNA5 ISA §8.3/8.4: g0 = count=1 | lds_addr | global_addr | type=2
// g1 = data_size=8B, tensor_dim0 = tile_dim0 = bytes/8, tile_dim1=0 (1-D).
// Issue from one wave; TENSORcnt tracks completion (in-order per wave).
// ---------------------------------------------------------------------------
__device__ __forceinline__ void tdm_load_1d(unsigned int lds_addr, const void* gptr,
                                            unsigned int bytes) {
  unsigned long long ga = (unsigned long long)gptr;
  unsigned int n = bytes >> 3;              // 8-byte elements
  v4u g0;
  g0[0] = 1u;                                        // count = 1 (valid descriptor)
  g0[1] = lds_addr;                                  // LDS byte address
  g0[2] = (unsigned int)ga;                          // global_addr[31:0]
  g0[3] = ((unsigned int)(ga >> 32) & 0x01FFFFFFu)   // global_addr[56:32]
          | 0x80000000u;                             // type = 2 ("image")
  v8u g1;
  g1[0] = 3u << 16;                 // workgroup_mask=0, data_size=3 (8B), no flags
  g1[1] = (n & 0xFFFFu) << 16;      // tensor_dim0[15:0]   (bits 63:48)
  g1[2] = (n >> 16) | (1u << 16);   // tensor_dim0[31:16] | tensor_dim1=1
  g1[3] = (n & 0xFFFFu) << 16;      // tile_dim0 (bits 127:112)
  g1[4] = 0u;                       // tile_dim1=0 (1-D), tile_dim2=0
  g1[5] = n;                        // tensor_dim0_stride (unused for 1-D)
  g1[6] = 0u;
  g1[7] = 0u;
  asm volatile("tensor_load_to_lds %0, %1" :: "s"(g0), "s"(g1) : "memory");
}

__device__ __forceinline__ unsigned int lds_addr_of(const void* p) {
  // Generic LDS pointers map to the LDS aperture; addr[31:0] is the LDS offset.
  return (unsigned int)(size_t)p;
}

// ---------------------------------------------------------------------------
// Split an fp32 array into bf16 hi/lo pair (bf16x3 GEMM trick)
// ---------------------------------------------------------------------------
__global__ __launch_bounds__(256) void k_split(const float* __restrict__ src,
                                               unsigned short* __restrict__ hi,
                                               unsigned short* __restrict__ lo, int n) {
  int i = blockIdx.x * 256 + threadIdx.x;
  if (i < n) {
    float v = src[i];
    unsigned short h = f2bf(v);
    hi[i] = h;
    lo[i] = f2bf(v - bf2f(h));
  }
}

// ---------------------------------------------------------------------------
// p4 [B][C][N] (channel-major) -> node-major bf16 hi/lo [B][N][C]
// LDS-tiled transpose, coalesced both directions.
// ---------------------------------------------------------------------------
__global__ __launch_bounds__(256) void k_transpose_split(const float* __restrict__ p4,
                                                         unsigned short* __restrict__ xhi,
                                                         unsigned short* __restrict__ xlo) {
  __shared__ float tile[64][65];
  const int b  = blockIdx.y;
  const int n0 = blockIdx.x * 64;
  const int t  = threadIdx.x;
  for (int c0 = 0; c0 < CDIM; c0 += 64) {
#pragma unroll
    for (int r = 0; r < 16; ++r) {
      int cc = (t >> 6) * 16 + r;
      int nn = t & 63;
      tile[cc][nn] = p4[((size_t)b * CDIM + (c0 + cc)) * NPTS + n0 + nn];
    }
    __syncthreads();
#pragma unroll
    for (int r = 0; r < 16; ++r) {
      int nn = (t >> 6) * 16 + r;
      int cc = t & 63;
      float v = tile[cc][nn];
      unsigned short h = f2bf(v);
      size_t o = ((size_t)b * NPTS + n0 + nn) * CDIM + c0 + cc;
      xhi[o] = h;
      xlo[o] = f2bf(v - bf2f(h));
    }
    __syncthreads();
  }
}

// ---------------------------------------------------------------------------
// Row squared norms (one wave32 per node, shuffle reduce)
// ---------------------------------------------------------------------------
__global__ __launch_bounds__(256) void k_sqnorm(const unsigned short* __restrict__ xhi,
                                                const unsigned short* __restrict__ xlo,
                                                float* __restrict__ sq) {
  int node = blockIdx.x * 8 + (threadIdx.x >> 5);
  int lane = threadIdx.x & 31;
  const unsigned short* ph = xhi + (size_t)node * CDIM;
  const unsigned short* pl = xlo + (size_t)node * CDIM;
  float s = 0.f;
#pragma unroll
  for (int i = 0; i < 8; ++i) {
    int c = lane + i * 32;
    float v = bf2f(ph[c]) + bf2f(pl[c]);
    s += v * v;
  }
#pragma unroll
  for (int off = 16; off >= 1; off >>= 1) s += __shfl_xor(s, off, 32);
  if (lane == 0) sq[node] = s;
}

// ---------------------------------------------------------------------------
// Fused distance GEMM (bf16x3 WMMA) + per-row top-8 selection.
// 128 threads (4 waves) handle 16 rows; 64-column chunks.
// TDM software pipeline with two LDS buffers:
//   issue TDM(lo->L) -> phase1 WMMA on H -> wait,barrier ->
//   issue TDM(hi(c+1)->H) -> phase2 WMMA on L + epilogue -> barrier ->
//   selection -> wait,barrier.
// Every DMA overlaps a compute phase; each wait has exactly one TDM in flight.
// distS and the final merge buffers share one LDS region (disjoint lifetimes).
// ---------------------------------------------------------------------------
__global__ __launch_bounds__(128) void k_knn(const unsigned short* __restrict__ xhi,
                                             const unsigned short* __restrict__ xlo,
                                             const float* __restrict__ sq,
                                             int* __restrict__ idx_out) {
  __shared__ unsigned int BstH[64 * 128];  // 32 KB: hi split of 64 cols x 256 ch
  __shared__ unsigned int BstL[64 * 128];  // 32 KB: lo split
  __shared__ alignas(16) unsigned char uS[8448];  // distS[16][66] f32 / pd+pi overlay
  __shared__ float sqS[64];

  float (*distS)[66] = (float (*)[66])uS;

  const int b    = blockIdx.y;
  const int row0 = blockIdx.x * 16;
  const int t    = threadIdx.x;
  const int lane = t & 31;
  const int wave = t >> 5;
  const int m16  = lane & 15;
  const int half = lane >> 4;
  const unsigned int hAddr = lds_addr_of(BstH);
  const unsigned int lAddr = lds_addr_of(BstL);

  // Preload A fragments (16 rows, full K=256, hi & lo splits) into registers.
  Frag Ah[8], Al[8];
  {
    const unsigned short* rh = xhi + ((size_t)b * NPTS + row0 + m16) * CDIM;
    const unsigned short* rl = xlo + ((size_t)b * NPTS + row0 + m16) * CDIM;
#pragma unroll
    for (int ks = 0; ks < 8; ++ks)
#pragma unroll
      for (int v = 0; v < 8; ++v) {
        int k = ks * 32 + half * 16 + v * 2;
        Ah[ks].u[v] = *(const unsigned int*)(rh + k);
        Al[ks].u[v] = *(const unsigned int*)(rl + k);
      }
  }

  // Per-thread partial top-8 (thread t: row t>>3, column slice t&7)
  float bd[KNN]; int bi[KNN];
#pragma unroll
  for (int q = 0; q < KNN; ++q) { bd[q] = 3.0e38f; bi[q] = 0; }
  const int selRow = t >> 3;
  const int selSub = t & 7;
  const int colL   = wave * 16 + m16;

  // prologue: fetch hi split of chunk 0
  if (wave == 0) {
    tdm_load_1d(hAddr, xhi + (size_t)b * NPTS * CDIM, 64 * CDIM * 2);
    __builtin_amdgcn_s_wait_tensorcnt(0);
  }
  __syncthreads();

  for (int col0 = 0; col0 < NPTS; col0 += 64) {
    if (t < 64) sqS[t] = sq[(size_t)b * NPTS + col0 + t];
    if (wave == 0)  // DMA lo(c) while phase-1 computes on H
      tdm_load_1d(lAddr, xlo + ((size_t)b * NPTS + col0) * CDIM, 64 * CDIM * 2);

    v8f acc = {0.f, 0.f, 0.f, 0.f, 0.f, 0.f, 0.f, 0.f};
#pragma unroll
    for (int ks = 0; ks < 8; ++ks) {       // phase 1: hi*hi + lo*hi
      Frag bf;
#pragma unroll
      for (int v = 0; v < 8; ++v) bf.u[v] = BstH[colL * 128 + ks * 16 + half * 8 + v];
      acc = WMMA_BF16(Ah[ks], bf, acc);
      acc = WMMA_BF16(Al[ks], bf, acc);
    }
    if (wave == 0) __builtin_amdgcn_s_wait_tensorcnt(0);  // lo(c) arrived
    __syncthreads();                       // B1: H free, L visible

    if (wave == 0 && col0 + 64 < NPTS)     // DMA hi(c+1) while phase-2 computes on L
      tdm_load_1d(hAddr, xhi + ((size_t)b * NPTS + col0 + 64) * CDIM, 64 * CDIM * 2);

#pragma unroll
    for (int ks = 0; ks < 8; ++ks) {       // phase 2: hi*lo
      Frag bf;
#pragma unroll
      for (int v = 0; v < 8; ++v) bf.u[v] = BstL[colL * 128 + ks * 16 + half * 8 + v];
      acc = WMMA_BF16(Ah[ks], bf, acc);
    }
    { // epilogue: comparable distance tile; exclude self
#pragma unroll
      for (int r = 0; r < 8; ++r) {
        int m = r + half * 8;
        float d = sqS[colL] - 2.0f * acc[r];
        if (col0 + colL == row0 + m) d = 3.0e38f;
        distS[m][colL] = d;
      }
    }
    __syncthreads();                       // B2: distS ready

    // partial top-8 update (static-index bubble insert: no scratch spills)
#pragma unroll
    for (int cc = 0; cc < 8; ++cc) {
      float d = distS[selRow][selSub * 8 + cc];
      if (d < bd[KNN - 1]) {
        bd[KNN - 1] = d; bi[KNN - 1] = col0 + selSub * 8 + cc;
#pragma unroll
        for (int p = KNN - 1; p > 0; --p)
          if (bd[p] < bd[p - 1]) {
            float td = bd[p]; bd[p] = bd[p - 1]; bd[p - 1] = td;
            int ti = bi[p]; bi[p] = bi[p - 1]; bi[p - 1] = ti;
          }
      }
    }
    if (wave == 0) __builtin_amdgcn_s_wait_tensorcnt(0);  // hi(c+1) arrived
    __syncthreads();                       // B3: end of chunk
  }

  // merge 8 partial lists per row (overlay region: distS is dead now)
  float* pdp = (float*)uS;                 // [16][8][8]
  int*   pip = (int*)(uS + 4096);          // [16][8][8]
#pragma unroll
  for (int q = 0; q < KNN; ++q) {
    pdp[(selRow * 8 + selSub) * 8 + q] = bd[q];
    pip[(selRow * 8 + selSub) * 8 + q] = bi[q];
  }
  __syncthreads();
  if (t < 16) {
    float fd[KNN]; int fi[KNN];
#pragma unroll
    for (int q = 0; q < KNN; ++q) { fd[q] = 3.0e38f; fi[q] = 0; }
    for (int s = 0; s < 8; ++s)
#pragma unroll
      for (int q = 0; q < KNN; ++q) {
        float d = pdp[(t * 8 + s) * 8 + q];
        if (d < fd[KNN - 1]) {
          fd[KNN - 1] = d; fi[KNN - 1] = pip[(t * 8 + s) * 8 + q];
#pragma unroll
          for (int p = KNN - 1; p > 0; --p)
            if (fd[p] < fd[p - 1]) {
              float td = fd[p]; fd[p] = fd[p - 1]; fd[p - 1] = td;
              int ti = fi[p]; fi[p] = fi[p - 1]; fi[p - 1] = ti;
            }
        }
      }
#pragma unroll
    for (int q = 0; q < KNN; ++q)
      idx_out[((size_t)b * NPTS + row0 + t) * KNN + q] = fi[q];
  }
}

// ---------------------------------------------------------------------------
// Mean-aggregate K=8 neighbors (reconstruct fp32 from hi+lo), re-split bf16.
// One block per node; thread == channel (coalesced gathers).
// ---------------------------------------------------------------------------
__global__ __launch_bounds__(256) void k_aggr(const unsigned short* __restrict__ shi,
                                              const unsigned short* __restrict__ slo,
                                              const int* __restrict__ idx,
                                              unsigned short* __restrict__ ohi,
                                              unsigned short* __restrict__ olo) {
  size_t node = blockIdx.x;                 // b*NPTS + n
  size_t b    = node / NPTS;
  int c = threadIdx.x;
  const int* row = idx + node * KNN;
  float s = 0.f;
#pragma unroll
  for (int j = 0; j < KNN; ++j) {
    size_t src = (b * NPTS + (size_t)row[j]) * CDIM + c;
    s += bf2f(shi[src]) + bf2f(slo[src]);
  }
  s *= (1.0f / KNN);
  unsigned short h = f2bf(s);
  ohi[node * CDIM + c] = h;
  olo[node * CDIM + c] = f2bf(s - bf2f(h));
}

// ---------------------------------------------------------------------------
// SAGEConv GEMM: out = aggr @ W_l^T + self @ W_r^T + bias  (optional ReLU)
// bf16x3 split on both operands: 6 WMMA per k-step per wave.
// Block 128 threads: 16 nodes x 64 outputs (4 waves x 16-col tiles).
// A tiles staged by 4 TDM DMAs (8 KB each).
// ---------------------------------------------------------------------------
__global__ __launch_bounds__(128) void k_sage_gemm(
    const unsigned short* __restrict__ Agh, const unsigned short* __restrict__ Agl,
    const unsigned short* __restrict__ Axh, const unsigned short* __restrict__ Axl,
    const unsigned short* __restrict__ Wlh, const unsigned short* __restrict__ Wll,
    const unsigned short* __restrict__ Wrh, const unsigned short* __restrict__ Wrl,
    const float* __restrict__ bias,
    float* __restrict__ out_f32, unsigned short* __restrict__ out_hi,
    unsigned short* __restrict__ out_lo, int Oc, int relu) {
  __shared__ unsigned int As[4][16 * 128];  // {aggr_hi, aggr_lo, self_hi, self_lo}
  const int b    = blockIdx.z;
  const int row0 = blockIdx.x * 16;
  const int o0   = blockIdx.y * 64;
  const int t    = threadIdx.x;
  const int lane = t & 31;
  const int wave = t >> 5;

  if (wave == 0) {
    const size_t rowOff = ((size_t)b * NPTS + row0) * CDIM;
    const unsigned int a0 = lds_addr_of(As);
    tdm_load_1d(a0 +     0, Agh + rowOff, 16 * CDIM * 2);
    tdm_load_1d(a0 +  8192, Agl + rowOff, 16 * CDIM * 2);
    tdm_load_1d(a0 + 16384, Axh + rowOff, 16 * CDIM * 2);
    tdm_load_1d(a0 + 24576, Axl + rowOff, 16 * CDIM * 2);
    __builtin_amdgcn_s_wait_tensorcnt(0);
  }
  __syncthreads();

  const int nl   = lane & 15;
  const int half = lane >> 4;
  const int oCol = o0 + wave * 16 + nl;
  v8f acc = {0.f, 0.f, 0.f, 0.f, 0.f, 0.f, 0.f, 0.f};
#pragma unroll
  for (int ks = 0; ks < 8; ++ks) {
    Frag a1h, a1l, a2h, a2l, w1h, w1l, w2h, w2l;
#pragma unroll
    for (int v = 0; v < 8; ++v) {
      int au = nl * 128 + ks * 16 + half * 8 + v;
      a1h.u[v] = As[0][au]; a1l.u[v] = As[1][au];
      a2h.u[v] = As[2][au]; a2l.u[v] = As[3][au];
      size_t wk = (size_t)oCol * CDIM + ks * 32 + half * 16 + v * 2;
      w1h.u[v] = *(const unsigned int*)(Wlh + wk);
      w1l.u[v] = *(const unsigned int*)(Wll + wk);
      w2h.u[v] = *(const unsigned int*)(Wrh + wk);
      w2l.u[v] = *(const unsigned int*)(Wrl + wk);
    }
    acc = WMMA_BF16(a1h, w1h, acc);
    acc = WMMA_BF16(a1l, w1h, acc);
    acc = WMMA_BF16(a1h, w1l, acc);
    acc = WMMA_BF16(a2h, w2h, acc);
    acc = WMMA_BF16(a2l, w2h, acc);
    acc = WMMA_BF16(a2h, w2l, acc);
  }
  float bv = bias[oCol];
#pragma unroll
  for (int r = 0; r < 8; ++r) {
    int m = r + half * 8;
    float val = acc[r] + bv;
    if (relu) val = fmaxf(val, 0.f);
    size_t o = ((size_t)b * NPTS + row0 + m) * Oc + oCol;
    if (out_f32) out_f32[o] = val;
    if (out_hi) {
      unsigned short h = f2bf(val);
      out_hi[o] = h;
      out_lo[o] = f2bf(val - bf2f(h));
    }
  }
}

// ---------------------------------------------------------------------------
extern "C" void kernel_launch(void* const* d_in, const int* in_sizes, int n_in,
                              void* d_out, int out_size, void* d_ws, size_t ws_size,
                              hipStream_t stream) {
  (void)in_sizes; (void)n_in; (void)out_size; (void)ws_size;
  const float* p4  = (const float*)d_in[0];
  const float* W1l = (const float*)d_in[1];
  const float* b1  = (const float*)d_in[2];
  const float* W1r = (const float*)d_in[3];
  const float* W2l = (const float*)d_in[4];
  const float* b2  = (const float*)d_in[5];
  const float* W2r = (const float*)d_in[6];

  const size_t BN = (size_t)BATCH * NPTS;
  char* ws = (char*)d_ws;
  size_t off = 0;
  auto take = [&](size_t bytes) -> void* {
    void* p = ws + off;
    off += (bytes + 255) & ~(size_t)255;
    return p;
  };
  unsigned short* xhi  = (unsigned short*)take(BN * CDIM * 2);
  unsigned short* xlo  = (unsigned short*)take(BN * CDIM * 2);
  unsigned short* aghi = (unsigned short*)take(BN * CDIM * 2);  // reused for layer2 aggr
  unsigned short* aglo = (unsigned short*)take(BN * CDIM * 2);
  unsigned short* hhi  = (unsigned short*)take(BN * HDIM * 2);
  unsigned short* hlo  = (unsigned short*)take(BN * HDIM * 2);
  float*          sq   = (float*)take(BN * 4);
  int*            idx  = (int*)take(BN * KNN * 4);
  unsigned short* w1lh = (unsigned short*)take(HDIM * CDIM * 2);
  unsigned short* w1ll = (unsigned short*)take(HDIM * CDIM * 2);
  unsigned short* w1rh = (unsigned short*)take(HDIM * CDIM * 2);
  unsigned short* w1rl = (unsigned short*)take(HDIM * CDIM * 2);
  unsigned short* w2lh = (unsigned short*)take(ODIM * HDIM * 2);
  unsigned short* w2ll = (unsigned short*)take(ODIM * HDIM * 2);
  unsigned short* w2rh = (unsigned short*)take(ODIM * HDIM * 2);
  unsigned short* w2rl = (unsigned short*)take(ODIM * HDIM * 2);

  // 1) weight splits
  k_split<<<(HDIM * CDIM + 255) / 256, 256, 0, stream>>>(W1l, w1lh, w1ll, HDIM * CDIM);
  k_split<<<(HDIM * CDIM + 255) / 256, 256, 0, stream>>>(W1r, w1rh, w1rl, HDIM * CDIM);
  k_split<<<(ODIM * HDIM + 255) / 256, 256, 0, stream>>>(W2l, w2lh, w2ll, ODIM * HDIM);
  k_split<<<(ODIM * HDIM + 255) / 256, 256, 0, stream>>>(W2r, w2rh, w2rl, ODIM * HDIM);

  // 2) transpose + split x
  k_transpose_split<<<dim3(NPTS / 64, BATCH), 256, 0, stream>>>(p4, xhi, xlo);

  // 3) row norms
  k_sqnorm<<<(unsigned)(BN / 8), 256, 0, stream>>>(xhi, xlo, sq);

  // 4) fused distance GEMM + top-8 (TDM-pipelined)
  k_knn<<<dim3(NPTS / 16, BATCH), 128, 0, stream>>>(xhi, xlo, sq, idx);

  // 5) layer-1 aggregation + SAGE GEMM (relu, emits bf16 split h)
  k_aggr<<<(unsigned)BN, 256, 0, stream>>>(xhi, xlo, idx, aghi, aglo);
  k_sage_gemm<<<dim3(NPTS / 16, HDIM / 64, BATCH), 128, 0, stream>>>(
      aghi, aglo, xhi, xlo, w1lh, w1ll, w1rh, w1rl, b1,
      nullptr, hhi, hlo, HDIM, 1);

  // 6) layer-2 aggregation + SAGE GEMM (fp32 out -> d_out)
  k_aggr<<<(unsigned)BN, 256, 0, stream>>>(hhi, hlo, idx, aghi, aglo);
  k_sage_gemm<<<dim3(NPTS / 16, ODIM / 64, BATCH), 128, 0, stream>>>(
      aghi, aglo, hhi, hlo, w2lh, w2ll, w2rh, w2rl, b2,
      (float*)d_out, nullptr, nullptr, ODIM, 0);
}
